// LSTM_38448547233933
// MI455X (gfx1250) — matrix operand order repaired and checked by reading, not compiled, not run on hardware
//
#include <hip/hip_runtime.h>

typedef __attribute__((ext_vector_type(16))) _Float16 v16h;
typedef __attribute__((ext_vector_type(8)))  _Float16 v8h;
typedef __attribute__((ext_vector_type(8)))  float    v8f;

#define HIDN      256
#define G4H       1024
#define BATCHN    256
#define INLEN     512
#define NSAMP     16
#define HORIZ     64
#define RB        16      // batch rows per workgroup
#define SA        264     // f16 element stride of LDS h arrays (pad vs 256)
#define SG        1028    // f32 element stride of LDS gates (pad vs 1024)

// ---- LDS carve offsets (bytes) ----
#define OFF_GATES 0
#define OFF_HA0   65792      // 16*1028*4
#define OFF_HA1   74240      // +16*264*2
#define OFF_C0    82688      // +16*264*2
#define OFF_C1    99072      // +16*256*4
#define OFF_IN    115456     // +16*256*4
#define OFF_PP    115520     // +16*4
#define SMEM_BYTES 116544    // +256*4

// Opaque zero offset, re-materialized every loop iteration: blocks LICM from
// hoisting the loop-invariant weight loads (which caused scratch spills), while
// keeping the pointer's addrspace(1) provenance so loads stay global_load (not flat).
__device__ __forceinline__ const _Float16* launder(const _Float16* p) {
  int ofs = 0;
  asm volatile("" : "+s"(ofs));
  return p + ofs;
}

// Hardware tanh (TRANS op on gfx1250) if the builtin exists; sigmoid via tanh.
#if __has_builtin(__builtin_amdgcn_tanhf)
__device__ __forceinline__ float hw_tanh(float x) { return __builtin_amdgcn_tanhf(x); }
#elif __has_builtin(__builtin_amdgcn_tanh_f32)
__device__ __forceinline__ float hw_tanh(float x) { return __builtin_amdgcn_tanh_f32(x); }
#else
__device__ __forceinline__ float hw_tanh(float x) { return tanhf(x); }
#endif
__device__ __forceinline__ float sigf(float x) { return 0.5f + 0.5f * hw_tanh(0.5f * x); }

// A fragment for v_wmma_f32_16x16x32_f16, contiguous k order (same as B in gemm8):
// lane l: mA = l&15, hi = l>>4; halves j = K(hi*16+j), j = 0..15
__device__ __forceinline__ v16h load_a_frag(const _Float16* hA, int mA, int hi, int k0) {
  const _Float16* p = hA + mA * SA + k0 + hi * 16;
  v8h lo = *(const v8h*)(p);
  v8h hv = *(const v8h*)(p + 8);
  v16h a;
#pragma unroll
  for (int i = 0; i < 8; ++i) { a[i] = lo[i]; a[i + 8] = hv[i]; }
  return a;
}

// One wave computes a 16-row x 128-col slab: 8 n-tiles, K=256 (8 k-steps).
// B in the same contiguous k order as A: lane l: n = n0+nt*16+(l&15), halves j = W[n][k0+hi*16+j]
// (contiguous 32B from row n of the f16 [1024,256] weight matrix).
__device__ __forceinline__ void gemm8(v8f acc[8], const _Float16* hA,
                                      const _Float16* W,
                                      int n0, int ln, int hi) {
  const _Float16* Wb = W + (size_t)(n0 + ln) * HIDN + hi * 16;
#pragma unroll
  for (int kk = 0; kk < 8; ++kk) {
    v16h b[8];
#pragma unroll
    for (int nt = 0; nt < 8; ++nt)
      b[nt] = *(const v16h*)(Wb + nt * 16 * HIDN + kk * 32);   // immediate offsets
    v16h a = load_a_frag(hA, ln, hi, kk * 32);
#pragma unroll
    for (int nt = 0; nt < 8; ++nt)
      acc[nt] = __builtin_amdgcn_wmma_f32_16x16x32_f16(
          false, a, false, b[nt], (short)0, acc[nt], false, false);
  }
}

// Epilogue: add rank-1 input term + bias, scatter C (m = hi*8+v, n = lane col) to LDS gates.
__device__ __forceinline__ void epi_store(v8f acc[8], float* gates, int n0, int ln, int hi,
                                          const float* inbuf,
                                          const float* __restrict__ Wih,
                                          const float* __restrict__ bias) {
#pragma unroll
  for (int nt = 0; nt < 8; ++nt) {
    int n = n0 + nt * 16 + ln;
    float wn = (Wih != nullptr) ? Wih[n] : 0.f;
    float bn = bias[n];
#pragma unroll
    for (int v = 0; v < 8; ++v) {
      int m = hi * 8 + v;
      float val = acc[nt][v] + bn;
      if (Wih != nullptr) val += inbuf[m] * wn;
      gates[m * SG + n] = val;
    }
  }
}

// i,f,g,o at column offsets 0/256/512/768 (PyTorch gate order). Each thread owns the
// same 16 (m,nh) elements it will own every step -> c state stays thread-consistent.
__device__ __forceinline__ void lstm_act(const float* gates, float* c, _Float16* hOut, int t) {
  int m = t >> 4, base = t & 15;
#pragma unroll 4
  for (int e = 0; e < 16; ++e) {
    int nh = base + e * 16;
    float ig = sigf(gates[m * SG + nh]);
    float fg = sigf(gates[m * SG + HIDN + nh]);
    float gg = hw_tanh(gates[m * SG + 2 * HIDN + nh]);
    float og = sigf(gates[m * SG + 3 * HIDN + nh]);
    float cn = fg * c[m * HIDN + nh] + ig * gg;
    c[m * HIDN + nh] = cn;
    hOut[m * SA + nh] = (_Float16)(og * hw_tanh(cn));
  }
}

// ---- f32 -> f16 weight conversion (6 matrices of 1024x256) ----
__global__ void __launch_bounds__(256) convert_weights(
    const float* __restrict__ s0, const float* __restrict__ s1,
    const float* __restrict__ s2, const float* __restrict__ s3,
    const float* __restrict__ s4, const float* __restrict__ s5,
    _Float16* __restrict__ dst) {
  int i = blockIdx.x * 256 + threadIdx.x;
  int which = i >> 18;            // 262144 elements per matrix
  int off = i & 262143;
  const float* s = (which == 0) ? s0 : (which == 1) ? s1 : (which == 2) ? s2
                 : (which == 3) ? s3 : (which == 4) ? s4 : s5;
  dst[i] = (_Float16)s[off];
}

// ---- Encoder: 16 persistent WGs x 16 rows; 512 sequential steps ----
__global__ void __launch_bounds__(256) lstm_encoder(
    const float* __restrict__ x,
    const _Float16* __restrict__ Whh0p, const _Float16* __restrict__ Wih1p,
    const _Float16* __restrict__ Whh1p,
    const float* __restrict__ Wih0, const float* __restrict__ b0,
    const float* __restrict__ b1,
    _Float16* __restrict__ eh0, float* __restrict__ ec0,
    _Float16* __restrict__ eh1, float* __restrict__ ec1) {
  extern __shared__ char smem[];
  float*    gates = (float*)(smem + OFF_GATES);
  _Float16* hA0   = (_Float16*)(smem + OFF_HA0);
  _Float16* hA1   = (_Float16*)(smem + OFF_HA1);
  float*    c0    = (float*)(smem + OFF_C0);
  float*    c1    = (float*)(smem + OFF_C1);
  float*    inbuf = (float*)(smem + OFF_IN);

  int t = threadIdx.x;
  int ln = t & 15, hi = (t >> 4) & 1, wave = t >> 5;
  int n0 = wave * 128;
  int row0 = blockIdx.x * RB;

  { // zero-init state
    int m = t >> 4, base = t & 15;
#pragma unroll 4
    for (int e = 0; e < 16; ++e) {
      int nh = base + e * 16;
      c0[m * HIDN + nh] = 0.f; c1[m * HIDN + nh] = 0.f;
      hA0[m * SA + nh] = (_Float16)0.f; hA1[m * SA + nh] = (_Float16)0.f;
    }
  }
  __syncthreads();

  for (int step = 0; step < INLEN; ++step) {
    const _Float16* Whh0 = launder(Whh0p);
    const _Float16* Wih1 = launder(Wih1p);
    const _Float16* Whh1 = launder(Whh1p);
    if (t < RB) inbuf[t] = x[(row0 + t) * INLEN + step];
    __syncthreads();
    { v8f acc[8] = {};
      gemm8(acc, hA0, Whh0, n0, ln, hi);
      epi_store(acc, gates, n0, ln, hi, inbuf, Wih0, b0); }
    __syncthreads();
    lstm_act(gates, c0, hA0, t);
    __syncthreads();
    { v8f acc[8] = {};
      gemm8(acc, hA0, Wih1, n0, ln, hi);
      gemm8(acc, hA1, Whh1, n0, ln, hi);
      epi_store(acc, gates, n0, ln, hi, nullptr, nullptr, b1); }
    __syncthreads();
    lstm_act(gates, c1, hA1, t);
    __syncthreads();
  }

  { // write final states to workspace
    int m = t >> 4, base = t & 15;
#pragma unroll 4
    for (int e = 0; e < 16; ++e) {
      int nh = base + e * 16;
      int g = (row0 + m) * HIDN + nh;
      eh0[g] = hA0[m * SA + nh]; ec0[g] = c0[m * HIDN + nh];
      eh1[g] = hA1[m * SA + nh]; ec1[g] = c1[m * HIDN + nh];
    }
  }
}

// ---- Decoder: 256 persistent WGs x 16 rows (S*B=4096); 64 sequential steps ----
__global__ void __launch_bounds__(256) lstm_decoder(
    const _Float16* __restrict__ Whh0p, const _Float16* __restrict__ Wih1p,
    const _Float16* __restrict__ Whh1p,
    const float* __restrict__ Wih0, const float* __restrict__ b0,
    const float* __restrict__ b1,
    const float* __restrict__ outW, const float* __restrict__ outB,
    const float* __restrict__ dec_init,
    const _Float16* __restrict__ eh0, const float* __restrict__ ec0,
    const _Float16* __restrict__ eh1, const float* __restrict__ ec1,
    float* __restrict__ out) {
  extern __shared__ char smem[];
  float*    gates = (float*)(smem + OFF_GATES);
  _Float16* hA0   = (_Float16*)(smem + OFF_HA0);
  _Float16* hA1   = (_Float16*)(smem + OFF_HA1);
  float*    c0    = (float*)(smem + OFF_C0);
  float*    c1    = (float*)(smem + OFF_C1);
  float*    inbuf = (float*)(smem + OFF_IN);
  float*    pp    = (float*)(smem + OFF_PP);

  int t = threadIdx.x;
  int ln = t & 15, hi = (t >> 4) & 1, wave = t >> 5;
  int n0 = wave * 128;
  int r0 = blockIdx.x * RB;

  { // init from encoder state, replicated across samples: row r -> batch b = r % 256
    int m = t >> 4, base = t & 15;
    int b = (r0 + m) & (BATCHN - 1);
#pragma unroll 4
    for (int e = 0; e < 16; ++e) {
      int nh = base + e * 16;
      hA0[m * SA + nh] = eh0[b * HIDN + nh]; c0[m * HIDN + nh] = ec0[b * HIDN + nh];
      hA1[m * SA + nh] = eh1[b * HIDN + nh]; c1[m * HIDN + nh] = ec1[b * HIDN + nh];
    }
  }
  if (t < RB) inbuf[t] = dec_init[r0 + t];
  __syncthreads();

  for (int step = 0; step < HORIZ; ++step) {
    const _Float16* Whh0 = launder(Whh0p);
    const _Float16* Wih1 = launder(Wih1p);
    const _Float16* Whh1 = launder(Whh1p);
    { v8f acc[8] = {};
      gemm8(acc, hA0, Whh0, n0, ln, hi);
      epi_store(acc, gates, n0, ln, hi, inbuf, Wih0, b0); }
    __syncthreads();
    lstm_act(gates, c0, hA0, t);
    __syncthreads();
    { v8f acc[8] = {};
      gemm8(acc, hA0, Wih1, n0, ln, hi);
      gemm8(acc, hA1, Whh1, n0, ln, hi);
      epi_store(acc, gates, n0, ln, hi, nullptr, nullptr, b1); }
    __syncthreads();
    lstm_act(gates, c1, hA1, t);
    __syncthreads();
    // output projection: pred[m] = h1[m,:] . outW + outB
    { int m = t >> 4, ks = t & 15;
      const _Float16* hr = hA1 + m * SA + ks * 16;
      const float* wr = outW + ks * 16;
      float p = 0.f;
#pragma unroll
      for (int k2 = 0; k2 < 16; ++k2) p += (float)hr[k2] * wr[k2];
      pp[t] = p; }
    __syncthreads();
    if (t < RB) {
      float p = outB[0];
#pragma unroll
      for (int k2 = 0; k2 < 16; ++k2) p += pp[t * 16 + k2];
      inbuf[t] = p;                               // feed back as next input
      int r = r0 + t;
      int b = r & (BATCHN - 1);
      int s = r >> 8;
      out[(size_t)b * (NSAMP * HORIZ) + s * HORIZ + step] = p;
    }
    __syncthreads();
  }
}

extern "C" void kernel_launch(void* const* d_in, const int* in_sizes, int n_in,
                              void* d_out, int out_size, void* d_ws, size_t ws_size,
                              hipStream_t stream) {
  const float* x        = (const float*)d_in[0];
  const float* enc_Wih0 = (const float*)d_in[1];
  const float* enc_Whh0 = (const float*)d_in[2];
  const float* enc_b0   = (const float*)d_in[3];
  const float* enc_Wih1 = (const float*)d_in[4];
  const float* enc_Whh1 = (const float*)d_in[5];
  const float* enc_b1   = (const float*)d_in[6];
  const float* dec_Wih0 = (const float*)d_in[7];
  const float* dec_Whh0 = (const float*)d_in[8];
  const float* dec_b0   = (const float*)d_in[9];
  const float* dec_Wih1 = (const float*)d_in[10];
  const float* dec_Whh1 = (const float*)d_in[11];
  const float* dec_b1   = (const float*)d_in[12];
  const float* out_W    = (const float*)d_in[13];
  const float* out_b    = (const float*)d_in[14];
  const float* dec_init = (const float*)d_in[15];

  char* ws = (char*)d_ws;
  _Float16* wt   = (_Float16*)ws;               // 6 x 1024x256 f16 weights
  _Float16* eWhh0 = wt + 0 * 262144;
  _Float16* eWih1 = wt + 1 * 262144;
  _Float16* eWhh1 = wt + 2 * 262144;
  _Float16* dWhh0 = wt + 3 * 262144;
  _Float16* dWih1 = wt + 4 * 262144;
  _Float16* dWhh1 = wt + 5 * 262144;
  _Float16* eh0 = (_Float16*)(ws + 3145728);    // encoder final states
  _Float16* eh1 = (_Float16*)(ws + 3276800);
  float*    ec0 = (float*)(ws + 3407872);
  float*    ec1 = (float*)(ws + 3670016);       // total ws use: 3,932,160 B

  convert_weights<<<6144, 256, 0, stream>>>(enc_Whh0, enc_Wih1, enc_Whh1,
                                            dec_Whh0, dec_Wih1, dec_Whh1, wt);
  lstm_encoder<<<16, 256, SMEM_BYTES, stream>>>(x, eWhh0, eWih1, eWhh1,
                                                enc_Wih0, enc_b0, enc_b1,
                                                eh0, ec0, eh1, ec1);
  lstm_decoder<<<256, 256, SMEM_BYTES, stream>>>(dWhh0, dWih1, dWhh1,
                                                 dec_Wih0, dec_b0, dec_b1,
                                                 out_W, out_b, dec_init,
                                                 eh0, ec0, eh1, ec1,
                                                 (float*)d_out);
}
